// SE3ConvolutionBN_16217796509737
// MI455X (gfx1250) — compile-verified, matrix operand order checked by reading, not run on hardware
//
#include <hip/hip_runtime.h>

typedef __attribute__((ext_vector_type(16))) __bf16 v16bf;
typedef __attribute__((ext_vector_type(8)))  __bf16 v8bf;
typedef __attribute__((ext_vector_type(8)))  float  v8f;

// ---------------- geometry ----------------
#define BATCH   4
#define CIN     72
#define CPAD    96          // ci padded to 3 chunks of 32 for K=32 bf16 WMMA
#define COTILES 5           // 5 tiles of 16 output channels (72 -> 80 padded)
#define DSP     32
#define XP      36          // 32 + 2*2 halo
#define TAPS    125
#define SPATIAL (DSP*DSP*DSP)          // 32768
#define NTOT    (BATCH*(size_t)SPATIAL) // 131072 per channel
#define OUTELEM ((size_t)BATCH*CIN*SPATIAL)

// workspace layout (bytes)
#define OFF_WP    0                       // packed weights: 5*125*3*512 bf16 = 1.92 MB
#define OFF_XB    (2u<<20)                // padded bf16 x: 4*36^3*96*2 = 34.2 MB
#define OFF_STATS (40u<<20)               // 144 floats (sum, sumsq per channel)
#define OFF_SC    (OFF_STATS + 4096)      // 72 floats
#define OFF_SH    (OFF_SC + 512)          // 72 floats

#define WBLK      512                     // bf16 elems per fragment block (32 lanes * 16)
#define WTSTRIDE  (TAPS*3*WBLK)           // per-co-tile stride in packed weights

static __device__ __forceinline__ unsigned short f2bf(float f) {
    unsigned int u = __builtin_bit_cast(unsigned int, f);
    unsigned int r = u + 0x7FFFu + ((u >> 16) & 1u);   // round-to-nearest-even
    return (unsigned short)(r >> 16);
}

// ---------------- 0: zero stats ----------------
__global__ void k_zero_stats(float* stats) {
    if (threadIdx.x < 144) stats[threadIdx.x] = 0.0f;
}

// ---------------- 1: pack weights into A-fragment order ----------------
// block (t, tap, c): 32 lanes * 16 bf16.  lane l -> M = l%16 (co = 16t+M)
// K(e): lane<16: e<8 ? e : e+8 ; lane>=16: e<8 ? e+8 : e+16 ; ci = 32c + K
__global__ void k_pack_w(const float* __restrict__ ker, unsigned short* __restrict__ wp) {
    int idx = blockIdx.x * blockDim.x + threadIdx.x;
    if (idx >= COTILES * TAPS * 3 * WBLK) return;
    int e   = idx & 15;
    int l   = (idx >> 4) & 31;
    int blk = idx >> 9;
    int c   = blk % 3;
    int tap = (blk / 3) % TAPS;
    int t   = blk / (3 * TAPS);
    int m   = l & 15;
    int co  = 16 * t + m;
    int K   = (l < 16) ? ((e < 8) ? e : e + 8) : ((e < 8) ? e + 8 : e + 16);
    int ci  = 32 * c + K;
    float v = 0.0f;
    if (co < CIN && ci < CIN)
        v = ker[((size_t)co * CIN + ci) * TAPS + tap];
    wp[idx] = f2bf(v);
}

// ---------------- 2: pad + convert x, channel-last [b][zp][yp][xp][96] ----------------
__global__ void k_pad_x(const float* __restrict__ x, unsigned short* __restrict__ xb) {
    size_t idx = (size_t)blockIdx.x * blockDim.x + threadIdx.x;
    size_t total = (size_t)BATCH * XP * XP * XP * CPAD;
    if (idx >= total) return;
    int ci = (int)(idx % CPAD);
    size_t r = idx / CPAD;
    int xp = (int)(r % XP); r /= XP;
    int yp = (int)(r % XP); r /= XP;
    int zp = (int)(r % XP);
    int b  = (int)(r / XP);
    float v = 0.0f;
    if (ci < CIN && zp >= 2 && zp < 34 && yp >= 2 && yp < 34 && xp >= 2 && xp < 34) {
        v = x[(((size_t)b * CIN + ci) * DSP + (zp - 2)) * DSP * DSP
              + (size_t)(yp - 2) * DSP + (xp - 2)];
    }
    xb[idx] = f2bf(v);
}

// ---------------- 3: implicit-GEMM conv via bf16 WMMA ----------------
// one wave = full W row (32 positions, 2 N-tiles) x all 80 (padded) out channels.
// A fragments amortized over both N-tiles: per tap-chunk 4 B-loads + 5 A-loads
// feed 10 WMMAs. 8 waves / 256-thread block; 4096 waves total.
__global__ void __launch_bounds__(256)
k_conv_wmma(const __bf16* __restrict__ xb, const __bf16* __restrict__ wp,
            float* __restrict__ out) {
    int wave = blockIdx.x * (blockDim.x >> 5) + (threadIdx.x >> 5);
    int lane = threadIdx.x & 31;
    int y  = wave & 31;
    int z  = (wave >> 5) & 31;
    int b  = wave >> 10;            // always 0..3 (grid sized exactly)
    int nIdx = lane & 15;
    int hi   = lane >> 4;

    v8f accA[COTILES];   // positions 0..15
    v8f accB[COTILES];   // positions 16..31
#pragma unroll
    for (int t = 0; t < COTILES; ++t) {
        accA[t] = (v8f){0,0,0,0,0,0,0,0};
        accB[t] = (v8f){0,0,0,0,0,0,0,0};
    }

#pragma unroll 1
    for (int kd = 0; kd < 5; ++kd) {
#pragma unroll 1
        for (int kh = 0; kh < 5; ++kh) {
            const __bf16* row = xb +
                ((((size_t)b * XP + (z + kd)) * XP + (y + kh)) * XP) * CPAD;
#pragma unroll 1
            for (int kw = 0; kw < 5; ++kw) {
                const __bf16* posA = row + (size_t)(nIdx + kw) * CPAD + (hi << 3);
                const __bf16* posB = posA + 16 * CPAD;
                int tap = (kd * 5 + kh) * 5 + kw;
                const __bf16* wb = wp + (size_t)tap * (3 * WBLK) + (lane << 4);
#pragma unroll
                for (int c = 0; c < 3; ++c) {
                    v8bf aLo = *(const v8bf*)(posA + 32 * c);
                    v8bf aHi = *(const v8bf*)(posA + 32 * c + 16);
                    v8bf bLo = *(const v8bf*)(posB + 32 * c);
                    v8bf bHi = *(const v8bf*)(posB + 32 * c + 16);
                    v16bf bmatA = __builtin_shufflevector(aLo, aHi,
                        0,1,2,3,4,5,6,7,8,9,10,11,12,13,14,15);
                    v16bf bmatB = __builtin_shufflevector(bLo, bHi,
                        0,1,2,3,4,5,6,7,8,9,10,11,12,13,14,15);
#pragma unroll
                    for (int t = 0; t < COTILES; ++t) {
                        v16bf amat = *(const v16bf*)(wb + (size_t)c * WBLK
                                                        + (size_t)t * WTSTRIDE);
                        accA[t] = __builtin_amdgcn_wmma_f32_16x16x32_bf16(
                            false, amat, false, bmatA, (short)0, accA[t], false, false);
                        accB[t] = __builtin_amdgcn_wmma_f32_16x16x32_bf16(
                            false, amat, false, bmatB, (short)0, accB[t], false, false);
                    }
                }
            }
        }
    }

    // writeback: D element (vgpr v, lane l): M = v + 8*hi, N = l%16
    size_t sp = ((size_t)z << 10) + ((size_t)y << 5) + nIdx;
#pragma unroll
    for (int t = 0; t < COTILES; ++t) {
#pragma unroll
        for (int v = 0; v < 8; ++v) {
            int co = 16 * t + v + (hi << 3);
            if (co < CIN) {
                size_t base = (((size_t)b * CIN + co) << 15) + sp;
                out[base]      = accA[t][v];
                out[base + 16] = accB[t][v];
            }
        }
    }
}

// ---------------- 4: per-channel sum / sumsq ----------------
__global__ void k_stats(const float* __restrict__ y, float* __restrict__ stats) {
    int c    = blockIdx.x >> 5;     // 72 channels
    int part = blockIdx.x & 31;     // 32 partials per channel
    int tid  = threadIdx.x;
    float s1 = 0.0f, s2 = 0.0f;
    for (int j = part * 256 + tid; j < (int)NTOT; j += 32 * 256) {
        int b = j >> 15;
        int s = j & 32767;
        float v = y[(((size_t)b * CIN + c) << 15) + s];
        s1 += v; s2 += v * v;
    }
#pragma unroll
    for (int off = 16; off > 0; off >>= 1) {
        s1 += __shfl_xor(s1, off, 32);
        s2 += __shfl_xor(s2, off, 32);
    }
    __shared__ float sm[8][2];
    int w = tid >> 5, l = tid & 31;
    if (l == 0) { sm[w][0] = s1; sm[w][1] = s2; }
    __syncthreads();
    if (tid == 0) {
        float a = 0.0f, q = 0.0f;
#pragma unroll
        for (int i = 0; i < 8; ++i) { a += sm[i][0]; q += sm[i][1]; }
        atomicAdd(&stats[2 * c],     a);
        atomicAdd(&stats[2 * c + 1], q);
    }
}

// ---------------- 5: fold RS structure into per-channel scale/shift ----------------
__global__ void k_factors(const float* __restrict__ stats,
                          const float* __restrict__ bnw, const float* __restrict__ bnb,
                          float* __restrict__ sc, float* __restrict__ sh) {
    int c = threadIdx.x;
    if (c >= CIN) return;
    const float invN = 1.0f / (float)NTOT;
    float scale, shift;
    if (c < 8) {                              // scalar fields: center + scale + bias
        float mean = stats[2 * c] * invN;
        float var  = stats[2 * c + 1] * invN - mean * mean;
        float f    = bnw[c] * rsqrtf(var + 1e-5f);
        scale = f;
        shift = bnb[c] - mean * f;
    } else if (c < 32) {                      // l=1 fields (d=3)
        int m  = (c - 8) / 3;
        int c0 = 8 + 3 * m;
        float n2 = (stats[2*c0+1] + stats[2*(c0+1)+1] + stats[2*(c0+2)+1]) * invN;
        scale = bnw[8 + m] * rsqrtf(n2 + 1e-5f);
        shift = 0.0f;
    } else {                                  // l=2 fields (d=5)
        int m  = (c - 32) / 5;
        int c0 = 32 + 5 * m;
        float n2 = 0.0f;
#pragma unroll
        for (int i = 0; i < 5; ++i) n2 += stats[2 * (c0 + i) + 1];
        n2 *= invN;
        scale = bnw[16 + m] * rsqrtf(n2 + 1e-5f);
        shift = 0.0f;
    }
    sc[c] = scale;
    sh[c] = shift;
}

// ---------------- 6: in-place normalize ----------------
__global__ void k_norm(float* __restrict__ y,
                       const float* __restrict__ sc, const float* __restrict__ sh) {
    size_t stride = (size_t)gridDim.x * blockDim.x;
    for (size_t i = (size_t)blockIdx.x * blockDim.x + threadIdx.x;
         i < OUTELEM; i += stride) {
        int c = (int)((i >> 15) % CIN);
        y[i] = y[i] * sc[c] + sh[c];
    }
}

extern "C" void kernel_launch(void* const* d_in, const int* in_sizes, int n_in,
                              void* d_out, int out_size, void* d_ws, size_t ws_size,
                              hipStream_t stream) {
    (void)in_sizes; (void)n_in; (void)out_size; (void)ws_size;
    const float* x   = (const float*)d_in[0];
    const float* ker = (const float*)d_in[1];
    const float* bnw = (const float*)d_in[2];
    const float* bnb = (const float*)d_in[3];
    float* out = (float*)d_out;
    char*  ws  = (char*)d_ws;

    unsigned short* wp_s = (unsigned short*)(ws + OFF_WP);
    unsigned short* xb_s = (unsigned short*)(ws + OFF_XB);
    float* stats = (float*)(ws + OFF_STATS);
    float* sc    = (float*)(ws + OFF_SC);
    float* sh    = (float*)(ws + OFF_SH);

    k_zero_stats<<<1, 256, 0, stream>>>(stats);

    int packElems = COTILES * TAPS * 3 * WBLK;                 // 960000
    k_pack_w<<<(packElems + 255) / 256, 256, 0, stream>>>(ker, wp_s);

    size_t padElems = (size_t)BATCH * XP * XP * XP * CPAD;     // 17,915,904
    k_pad_x<<<(unsigned)((padElems + 255) / 256), 256, 0, stream>>>(x, xb_s);

    // 4096 waves (one per W row per batch/z/y), 8 waves per 256-thread block
    k_conv_wmma<<<512, 256, 0, stream>>>((const __bf16*)xb_s, (const __bf16*)wp_s, out);

    k_stats<<<CIN * 32, 256, 0, stream>>>(out, stats);
    k_factors<<<1, 128, 0, stream>>>(stats, bnw, bnb, sc, sh);
    k_norm<<<4096, 256, 0, stream>>>(out, sc, sh);
}
